// SpectralDecoder_13134009991359
// MI455X (gfx1250) — matrix operand order repaired and checked by reading, not compiled, not run on hardware
//
#include <hip/hip_runtime.h>
#include <hip/hip_bf16.h>

typedef __attribute__((ext_vector_type(16))) _Float16 v16h;
typedef __attribute__((ext_vector_type(8)))  float    v8f;

#define TWO_PI_OMEGA 188.49555921538757f   // 2*pi*30
#define F_DIM 256
#define CIN   128
#define COUT  16
#define MT    128                           // spatial points per block tile

// fast silu: u * sigmoid(u) with v_rcp_f32 instead of IEEE divide expansion
__device__ __forceinline__ float fast_silu(float u) {
    return u * __builtin_amdgcn_rcpf(1.0f + __expf(-u));
}

// ---------------------------------------------------------------------------
// Kernel 1: H = silu(h @ Wh + bh)  -> f16, row-major (b, f, c)
// ---------------------------------------------------------------------------
__global__ __launch_bounds__(256) void h_gemm_silu(
    const float* __restrict__ h, const float* __restrict__ Wh,
    const float* __restrict__ bh, _Float16* __restrict__ Hhalf)
{
    int tid = blockIdx.x * blockDim.x + threadIdx.x;     // b*65536 + f*256 + c
    int c = tid & 255;
    int f = (tid >> 8) & 255;
    int b = tid >> 16;
    const float* hr = h + ((size_t)b * F_DIM + f) * CIN;
    float acc = bh[c];
    #pragma unroll 4
    for (int k = 0; k < CIN; ++k)
        acc = fmaf(hr[k], Wh[k * F_DIM + c], acc);
    Hhalf[(size_t)tid] = (_Float16)fast_silu(acc);
}

// ---------------------------------------------------------------------------
// Kernel 2: pack H and Wu into WMMA B-fragment layout.
// B frag as packed here (contiguous K order, not AMD's table): lane n = lane%16 is the N column, khi = (lane>=16)*16,
// half index hh -> K = khi + hh (VGPR v packs K=2v,2v+1). One contiguous
// 16-half (32B) run per lane -> single aligned v16h global load in the GEMM.
// Hpack[b][ks][ct][lane][16], Wupack[ks][lane][16]
// ---------------------------------------------------------------------------
__global__ __launch_bounds__(256) void pack_frags(
    const _Float16* __restrict__ Hhalf, const float* __restrict__ Wu,
    _Float16* __restrict__ Hpack, _Float16* __restrict__ Wupack, int nH)
{
    int t = blockIdx.x * blockDim.x + threadIdx.x;
    if (t < nH) {
        int lane = t & 31;
        int ct   = (t >> 5) & 15;
        int ks   = (t >> 9) & 7;
        int b    = t >> 12;
        int n    = lane & 15;
        int khi  = (lane >= 16) ? 16 : 0;
        const _Float16* src = Hhalf + ((size_t)b * F_DIM + ks * 32 + khi) * F_DIM + ct * 16 + n;
        _Float16* dst = Hpack + (size_t)t * 16;
        #pragma unroll
        for (int hh = 0; hh < 16; ++hh)
            dst[hh] = src[(size_t)hh * F_DIM];           // stride over K (rows of H)
    } else if (t < nH + 256) {
        int s    = t - nH;
        int lane = s & 31;
        int ks   = s >> 5;
        int n    = lane & 15;
        int khi  = (lane >= 16) ? 16 : 0;
        _Float16* dst = Wupack + (size_t)s * 16;
        #pragma unroll
        for (int hh = 0; hh < 16; ++hh)
            dst[hh] = (_Float16)Wu[(ks * 32 + khi + hh) * COUT + n];
    }
}

// ---------------------------------------------------------------------------
// Kernel 3: fused decoder. One block = 128 spatial points (MT=128).
//   stage A: S[128][256] = sin(2*pi*30*(y@Wy+by)) into LDS (f16), 64KB.
//   stage B: U = Y @ H / N via wmma f16. Each wave owns a 4x4 grid of 16x16
//            accumulators (64 rows x 64 cols): every A fragment AND every B
//            fragment feeds 4 WMMAs (8KB operand bytes per 16 WMMAs).
//   stage C: silu(U) restaged into the SAME LDS buffer (Y fully consumed);
//            out = U @ Wu + bu via wmma, one 16-row tile per wave (all 8).
// ---------------------------------------------------------------------------
__global__ __launch_bounds__(256) void fused_decoder(
    const float* __restrict__ y,  const float* __restrict__ Wy,
    const float* __restrict__ by, const _Float16* __restrict__ Hpack,
    const _Float16* __restrict__ Wupack, const float* __restrict__ bu,
    float* __restrict__ out, int npts, float scale)
{
    __shared__ _Float16 Slds[MT][F_DIM];     // 64KB: Y tile, then reused for silu(U)

    const int b    = blockIdx.y;
    const int m0   = blockIdx.x * MT;
    const int t    = threadIdx.x;
    const int lane = t & 31;
    const int wave = t >> 5;

    // ---- stage A: Fourier features into LDS ----
    {
        int r  = t >> 1;                 // 128 rows, 2 threads per row
        int k0 = (t & 1) * 128;          // each thread: 128 consecutive features
        const float* yr = y + ((size_t)b * npts + m0 + r) * 3;
        float y0 = yr[0], y1 = yr[1], y2 = yr[2];
        #pragma unroll 8
        for (int k = k0; k < k0 + 128; ++k) {
            float ph = fmaf(y0, Wy[k],
                       fmaf(y1, Wy[F_DIM + k],
                       fmaf(y2, Wy[2 * F_DIM + k], by[k])));
            Slds[r][k] = (_Float16)__sinf(TWO_PI_OMEGA * ph);
        }
    }
    __syncthreads();

    // ---- stage B: U = Y @ H (K = 256, 8 steps of 32) ----
    const int rt0  = (wave & 1) * 4;         // 4 row-tiles of 16 per wave
    const int ct0  = (wave >> 1) * 4;        // 4 col-tiles of 16 per wave
    const int klo  = (lane >> 4) * 16;       // contiguous K half offset (matches packed B)
    const int crow = (lane >> 4) * 8;        // C/D layout: M = crow + v
    const int ncol = lane & 15;
    const int arow = lane & 15;              // row within a 16-row tile

    const _Float16* Hb = Hpack + (size_t)b * (8 * 16 * 32 * 16);

    v8f acc[4][4] = {};                      // [row-tile][col-tile] 64x64 per wave
    for (int ks = 0; ks < 8; ++ks) {
        // A fragments for this wave's 4 row-tiles (from LDS):
        // lane holds row (rt0+rt)*16+arow, K = ks*32 + klo + {0..15}
        v16h a[4];
        #pragma unroll
        for (int rt = 0; rt < 4; ++rt) {
            const _Float16* ya = &Slds[(rt0 + rt) * 16 + arow][ks * 32 + klo];
            #pragma unroll
            for (int i = 0; i < 8; ++i) { a[rt][i] = ya[i]; a[rt][i + 8] = ya[i + 8]; }
        }
        // B fragments for this wave's 4 col-tiles; each feeds 4 WMMAs
        #pragma unroll
        for (int c = 0; c < 4; ++c) {
            const v16h bf = *(const v16h*)(Hb + (((size_t)(ks * 16 + ct0 + c)) * 32 + lane) * 16);
            #pragma unroll
            for (int rt = 0; rt < 4; ++rt)
                acc[rt][c] = __builtin_amdgcn_wmma_f32_16x16x32_f16(
                    false, a[rt], false, bf, (short)0, acc[rt][c], false, false);
        }
    }

    // Y tile fully consumed by all waves before overwriting with U
    __syncthreads();

    // ---- scale + silu, restage into LDS (C/D layout -> row-major f16) ----
    #pragma unroll
    for (int rt = 0; rt < 4; ++rt) {
        #pragma unroll
        for (int c = 0; c < 4; ++c) {
            #pragma unroll
            for (int v = 0; v < 8; ++v) {
                float u = fast_silu(acc[rt][c][v] * scale);
                Slds[(rt0 + rt) * 16 + crow + v][(ct0 + c) * 16 + ncol] = (_Float16)u;
            }
        }
    }
    __syncthreads();

    // ---- stage C: out = silu(U) @ Wu + bu  (one 16-row tile per wave) ----
    {
        const int row2 = wave * 16 + arow;
        v8f acc2 = {};
        for (int ks = 0; ks < 8; ++ks) {
            v16h a;
            const _Float16* ua = &Slds[row2][ks * 32 + klo];
            #pragma unroll
            for (int i = 0; i < 8; ++i) { a[i] = ua[i]; a[i + 8] = ua[i + 8]; }
            const v16h bf = *(const v16h*)(Wupack + ((size_t)(ks * 32 + lane)) * 16);
            acc2 = __builtin_amdgcn_wmma_f32_16x16x32_f16(
                false, a, false, bf, (short)0, acc2, false, false);
        }
        const float bias = bu[ncol];
        #pragma unroll
        for (int v = 0; v < 8; ++v) {
            int m = wave * 16 + crow + v;
            out[(((size_t)b * npts) + m0 + m) * COUT + ncol] = acc2[v] + bias;
        }
    }
}

// ---------------------------------------------------------------------------
// Launch
// ---------------------------------------------------------------------------
extern "C" void kernel_launch(void* const* d_in, const int* in_sizes, int n_in,
                              void* d_out, int out_size, void* d_ws, size_t ws_size,
                              hipStream_t stream) {
    const float* h  = (const float*)d_in[0];
    const float* y  = (const float*)d_in[1];
    const float* Wy = (const float*)d_in[2];
    const float* by = (const float*)d_in[3];
    const float* Wh = (const float*)d_in[4];
    const float* bh = (const float*)d_in[5];
    const float* Wu = (const float*)d_in[6];
    const float* bu = (const float*)d_in[7];

    const int B    = in_sizes[0] / (F_DIM * CIN);        // 2
    const int npts = in_sizes[1] / (B * 3);              // 48^3 = 110592

    // workspace layout
    _Float16* Hhalf  = (_Float16*)d_ws;                                  // B*256*256 f16
    _Float16* Hpack  = Hhalf + (size_t)B * F_DIM * F_DIM;                // B*8*16*32*16 f16
    _Float16* Wupack = Hpack + (size_t)B * 8 * 16 * 32 * 16;             // 8*32*16 f16

    const int n1 = B * F_DIM * F_DIM;
    h_gemm_silu<<<n1 / 256, 256, 0, stream>>>(h, Wh, bh, Hhalf);

    const int nH = B * 8 * 16 * 32;
    pack_frags<<<(nH + 256 + 255) / 256, 256, 0, stream>>>(Hhalf, Wu, Hpack, Wupack, nH);

    fused_decoder<<<dim3(npts / MT, B), 256, 0, stream>>>(
        y, Wy, by, Hpack, Wupack, bu, (float*)d_out, npts, 1.0f / (float)npts);
}
